// EntropyGatedAttention_74577812127861
// MI455X (gfx1250) — matrix operand order repaired and checked by reading, not compiled, not run on hardware
//
#include <hip/hip_runtime.h>

// ---------------------------------------------------------------------------
// Gated multi-head attention for MI455X (gfx1250, wave32, WMMA).
// B=2, S=2048, D_MODEL=1024, H=16, HEAD_DIM=64.
// bf16 WMMA (v_wmma_f32_16x16x32_bf16), f32 accumulate, flash-attention,
// async global->LDS staging (ASYNCcnt) for the GEMM A tiles.
// ---------------------------------------------------------------------------

typedef __attribute__((ext_vector_type(16))) __bf16 v16bf;
typedef __attribute__((ext_vector_type(8)))  __bf16 v8bf;
typedef __attribute__((ext_vector_type(4)))  __bf16 v4bf;
typedef __attribute__((ext_vector_type(8)))  float  v8f;

union f16x16u { v16bf v; v8bf h2[2]; };

#define D_MODEL  1024
#define N_HEADS  16
#define HEAD_DIM 64
#define BATCH    2
#define SEQ      2048
#define NTOK     (BATCH * SEQ)     // 4096 tokens

// ---------------------------------------------------------------------------
// fp32 -> bf16 bulk convert (vectorized x4)
// ---------------------------------------------------------------------------
__global__ void __launch_bounds__(256)
cvt_f32_bf16(const float* __restrict__ in, __bf16* __restrict__ out, int n4) {
  int i = blockIdx.x * 256 + threadIdx.x;
  if (i < n4) {
    float4 f = ((const float4*)in)[i];
    v4bf o;
    o[0] = (__bf16)f.x; o[1] = (__bf16)f.y;
    o[2] = (__bf16)f.z; o[3] = (__bf16)f.w;
    ((v4bf*)out)[i] = o;
  }
}

// ---------------------------------------------------------------------------
// Y[4096][1024] = X(bf16) @ W(bf16) + bias, fused epilogue:
//   mode 0: Yb = Y * gate                (bf16 out, QKV projections)
//   mode 1: Yf = x*(1-g) + Y*g           (f32 out, final blend)
// Block: 256 threads (8 waves). Block tile 256(M) x 64(N), K-step 32.
// Wave w owns rows [w*32, w*32+32) x 64 N cols -> 8 WMMA accumulators.
// A tile staged with GLOBAL_LOAD_ASYNC_TO_LDS_B128 (ASYNCcnt path).
// ---------------------------------------------------------------------------
__global__ void __launch_bounds__(256)
gemm_bf16_wmma(const __bf16* __restrict__ X, const __bf16* __restrict__ W,
               const float* __restrict__ bias, const int* __restrict__ gate,
               const float* __restrict__ xres,
               __bf16* __restrict__ Yb, float* __restrict__ Yf, int mode) {
  __shared__ __bf16 sA[256][32];   // X tile,  [m][k]            (16 KB)
  __shared__ __bf16 sB[64][32];    // W tile transposed [n][k]   (4 KB)

  const int tid  = threadIdx.x;
  const int wave = tid >> 5;
  const int lane = tid & 31;
  const int m16  = lane & 15;      // A-row / B-col / C-col per WMMA layout
  const int half = lane >> 4;      // lane group (0: lanes 0-15, 1: 16-31)
  const int bm   = blockIdx.y * 256;
  const int bn   = blockIdx.x * 64;

  v8f acc[2][4] = {};              // [m-subtile][n-subtile] 16x16 f32 tiles

  for (int k0 = 0; k0 < D_MODEL; k0 += 32) {
    // ---- stage A tile async: each thread owns one 32-elem (64 B) row.
    //      LDS address comes from the VDST operand; INST_OFFSET is added to
    //      both LDS and global addresses (ISA 10, async pseudocode).
    {
      const __bf16* asrc = X + (size_t)(bm + tid) * D_MODEL + k0;
      unsigned int la = (unsigned int)(uintptr_t)&sA[tid][0];   // LDS byte offset
      asm volatile(
          "global_load_async_to_lds_b128 %0, %1, off\n\t"
          "global_load_async_to_lds_b128 %0, %1, off offset:16\n\t"
          "global_load_async_to_lds_b128 %0, %1, off offset:32\n\t"
          "global_load_async_to_lds_b128 %0, %1, off offset:48"
          :: "v"(la), "v"(asrc) : "memory");
      if (k0 + 32 < D_MODEL)
        __builtin_prefetch(asrc + 32, 0, 3);     // global_prefetch next K tile
    }
    // ---- stage B tile transposed: read W row-major (coalesced on N),
    //      scatter into sB[n][k] so fragments are contiguous in k ----
    {
      int kk = tid >> 3;
      int nn = (tid & 7) * 8;
      const __bf16* src = W + (size_t)(k0 + kk) * D_MODEL + bn + nn;
      v8bf w = *(const v8bf*)src;
#pragma unroll
      for (int i = 0; i < 8; ++i) sB[nn + i][kk] = w[i];
    }
    asm volatile("s_wait_asynccnt 0" ::: "memory");   // async LDS writes done
    __syncthreads();

    // ---- A fragments (16x32 bf16 each): lane m16, e<8 -> k=half*8+e,
    //      e>=8 -> k=16+half*8+(e-8)  (ISA 7.12.2 16-bit A layout) ----
    f16x16u a[2];
#pragma unroll
    for (int mt = 0; mt < 2; ++mt) {
      const __bf16* ap = &sA[wave * 32 + mt * 16 + m16][0];
      a[mt].h2[0] = *(const v8bf*)(ap + half * 8);
      a[mt].h2[1] = *(const v8bf*)(ap + 16 + half * 8);
    }
#pragma unroll
    for (int nt = 0; nt < 4; ++nt) {
      // B fragment (32x16 bf16): lane col n=m16, e<8 -> k=half*8+e, e>=8 -> k=16+half*8+(e-8)
      f16x16u b;
      b.h2[0] = *(const v8bf*)&sB[nt * 16 + m16][half * 8];
      b.h2[1] = *(const v8bf*)&sB[nt * 16 + m16][16 + half * 8];
#pragma unroll
      for (int mt = 0; mt < 2; ++mt)
        acc[mt][nt] = __builtin_amdgcn_wmma_f32_16x16x32_bf16(
            false, a[mt].v, false, b.v, (short)0, acc[mt][nt], false, false);
    }
    __syncthreads();
  }

  // ---- epilogue: C layout col n=m16, row m = r + 8*half ----
#pragma unroll
  for (int mt = 0; mt < 2; ++mt) {
#pragma unroll
    for (int nt = 0; nt < 4; ++nt) {
      int gn = bn + nt * 16 + m16;
      float bv = bias[gn];
#pragma unroll
      for (int r = 0; r < 8; ++r) {
        int gm = bm + wave * 32 + mt * 16 + r + 8 * half;   // flat token index
        float y = acc[mt][nt][r] + bv;
        float g = (float)gate[gm];
        size_t idx = (size_t)gm * D_MODEL + gn;
        if (mode == 0) {
          Yb[idx] = (__bf16)(y * g);
        } else {
          Yf[idx] = xres[idx] * (1.0f - g) + y * g;
        }
      }
    }
  }
}

// ---------------------------------------------------------------------------
// Flash attention per (batch, head): softmax(Q K^T / 8) V, head_dim = 64.
// Grid: (SEQ/64, BATCH*N_HEADS). Block: 128 threads = 4 waves.
// Each wave owns 16 query rows; kv loop in tiles of 64 keys.
// ---------------------------------------------------------------------------
__global__ void __launch_bounds__(128)
flash_attn_wmma(const __bf16* __restrict__ Q, const __bf16* __restrict__ K,
                const __bf16* __restrict__ V, __bf16* __restrict__ Oout) {
  __shared__ __bf16 sP[4][16][64];   // per-wave P tile (bf16), 8 KB
  __shared__ __bf16 sV[64][64];      // V tile transposed [head_dim][kv], 8 KB

  const int tid  = threadIdx.x;
  const int wave = tid >> 5;
  const int lane = tid & 31;
  const int m16  = lane & 15;
  const int half = lane >> 4;
  const int bb   = blockIdx.y >> 4;          // batch
  const int h    = blockIdx.y & 15;          // head
  const int q0   = blockIdx.x * 64 + wave * 16;
  const size_t base = (size_t)bb * SEQ * D_MODEL + (size_t)h * HEAD_DIM;

  // ---- hoist Q A-fragments (two 32-wide head-dim chunks) ----
  f16x16u qf[2];
#pragma unroll
  for (int kc = 0; kc < 2; ++kc) {
    const __bf16* qp = Q + base + (size_t)(q0 + m16) * D_MODEL + kc * 32;
    qf[kc].h2[0] = *(const v8bf*)(qp + half * 8);
    qf[kc].h2[1] = *(const v8bf*)(qp + 16 + half * 8);
  }

  v8f o[4] = {};            // output accumulators, 16 rows x 64 head dims
  float rm[8], rl[8];       // running max / sum per row (r, half)
#pragma unroll
  for (int r = 0; r < 8; ++r) { rm[r] = -1e30f; rl[r] = 0.0f; }

  for (int kv0 = 0; kv0 < SEQ; kv0 += 64) {
    // ---- cooperative V stage, transposed: sV[col][kv_local] ----
    {
      int row = tid >> 1;                   // kv_local 0..63
      int cb  = (tid & 1) * 32;
      const __bf16* vp = V + base + (size_t)(kv0 + row) * D_MODEL + cb;
#pragma unroll
      for (int u = 0; u < 4; ++u) {
        v8bf vv = *(const v8bf*)(vp + u * 8);
#pragma unroll
        for (int i = 0; i < 8; ++i) sV[cb + u * 8 + i][row] = vv[i];
      }
    }

    // ---- scores: S = Q @ K^T   (4 key tiles x 2 head-dim chunks) ----
    v8f s[4] = {};
#pragma unroll
    for (int nt = 0; nt < 4; ++nt) {
#pragma unroll
      for (int kc = 0; kc < 2; ++kc) {
        // B fragment of K^T: lane col n = key (kv0+nt*16+m16),
        // k = head-dim kc*32 + half*8 + e (e<8), kc*32 + 16 + half*8 + (e-8) (e>=8)
        f16x16u bf;
        const __bf16* kp = K + base + (size_t)(kv0 + nt * 16 + m16) * D_MODEL
                             + kc * 32 + half * 8;
        bf.h2[0] = *(const v8bf*)kp;
        bf.h2[1] = *(const v8bf*)(kp + 16);
        s[nt] = __builtin_amdgcn_wmma_f32_16x16x32_bf16(
            false, qf[kc].v, false, bf.v, (short)0, s[nt], false, false);
      }
    }
#pragma unroll
    for (int nt = 0; nt < 4; ++nt)
#pragma unroll
      for (int r = 0; r < 8; ++r) s[nt][r] *= 0.125f;   // 1/sqrt(64)

    // ---- online softmax (rows live in 16-lane halves of the wave) ----
#pragma unroll
    for (int r = 0; r < 8; ++r) {
      float tmax = fmaxf(fmaxf(s[0][r], s[1][r]), fmaxf(s[2][r], s[3][r]));
#pragma unroll
      for (int msk = 1; msk < 16; msk <<= 1)
        tmax = fmaxf(tmax, __shfl_xor(tmax, msk, 32));
      float mnew  = fmaxf(rm[r], tmax);
      float scale = __expf(rm[r] - mnew);
      float psum  = 0.0f;
#pragma unroll
      for (int nt = 0; nt < 4; ++nt) {
        float p = __expf(s[nt][r] - mnew);
        psum += p;
        sP[wave][r + 8 * half][nt * 16 + m16] = (__bf16)p;
      }
#pragma unroll
      for (int msk = 1; msk < 16; msk <<= 1)
        psum += __shfl_xor(psum, msk, 32);
      rl[r] = rl[r] * scale + psum;
      rm[r] = mnew;
#pragma unroll
      for (int nt = 0; nt < 4; ++nt) o[nt][r] *= scale;
    }

    asm volatile("s_wait_dscnt 0" ::: "memory");  // LDS stores visible
    __syncthreads();                              // sV + sP ready

    // ---- O += P @ V ----
#pragma unroll
    for (int kc = 0; kc < 2; ++kc) {
      f16x16u af;
      af.h2[0] = *(const v8bf*)&sP[wave][m16][kc * 32 + half * 8];
      af.h2[1] = *(const v8bf*)&sP[wave][m16][kc * 32 + 16 + half * 8];
#pragma unroll
      for (int nt = 0; nt < 4; ++nt) {
        f16x16u bf;
        bf.h2[0] = *(const v8bf*)&sV[nt * 16 + m16][kc * 32 + half * 8];
        bf.h2[1] = *(const v8bf*)&sV[nt * 16 + m16][kc * 32 + 16 + half * 8];
        o[nt] = __builtin_amdgcn_wmma_f32_16x16x32_bf16(
            false, af.v, false, bf.v, (short)0, o[nt], false, false);
      }
    }
    __syncthreads();   // done reading this tile before restaging
  }

  // ---- normalize and write attention output (bf16) ----
#pragma unroll
  for (int nt = 0; nt < 4; ++nt) {
#pragma unroll
    for (int r = 0; r < 8; ++r) {
      int tok = q0 + r + 8 * half;
      float val = o[nt][r] / rl[r];
      Oout[base + (size_t)tok * D_MODEL + nt * 16 + m16] = (__bf16)val;
    }
  }
}

// ---------------------------------------------------------------------------
// launcher
// ---------------------------------------------------------------------------
extern "C" void kernel_launch(void* const* d_in, const int* in_sizes, int n_in,
                              void* d_out, int out_size, void* d_ws, size_t ws_size,
                              hipStream_t stream) {
  const float* x    = (const float*)d_in[0];
  const int*   gate = (const int*)d_in[1];
  const float* Wq = (const float*)d_in[2]; const float* bq = (const float*)d_in[3];
  const float* Wk = (const float*)d_in[4]; const float* bk = (const float*)d_in[5];
  const float* Wv = (const float*)d_in[6]; const float* bv = (const float*)d_in[7];
  const float* Wo = (const float*)d_in[8]; const float* bo = (const float*)d_in[9];
  float* out = (float*)d_out;

  // workspace layout (bf16 elements), total ~48 MB
  __bf16* ws  = (__bf16*)d_ws;
  __bf16* Xb  = ws;                                        // 4096x1024
  __bf16* Wqb = Xb  + (size_t)NTOK * D_MODEL;              // 1024x1024
  __bf16* Wkb = Wqb + (size_t)D_MODEL * D_MODEL;
  __bf16* Wvb = Wkb + (size_t)D_MODEL * D_MODEL;
  __bf16* Wob = Wvb + (size_t)D_MODEL * D_MODEL;
  __bf16* Qb  = Wob + (size_t)D_MODEL * D_MODEL;           // 4096x1024
  __bf16* Kb  = Qb  + (size_t)NTOK * D_MODEL;
  __bf16* Vb  = Kb  + (size_t)NTOK * D_MODEL;
  __bf16* Ab  = Vb  + (size_t)NTOK * D_MODEL;              // attn out

  const int nX4 = NTOK * D_MODEL / 4;
  const int nW4 = D_MODEL * D_MODEL / 4;
  cvt_f32_bf16<<<(nX4 + 255) / 256, 256, 0, stream>>>(x,  Xb,  nX4);
  cvt_f32_bf16<<<(nW4 + 255) / 256, 256, 0, stream>>>(Wq, Wqb, nW4);
  cvt_f32_bf16<<<(nW4 + 255) / 256, 256, 0, stream>>>(Wk, Wkb, nW4);
  cvt_f32_bf16<<<(nW4 + 255) / 256, 256, 0, stream>>>(Wv, Wvb, nW4);
  cvt_f32_bf16<<<(nW4 + 255) / 256, 256, 0, stream>>>(Wo, Wob, nW4);

  dim3 gg(D_MODEL / 64, NTOK / 256);   // (16, 16)
  gemm_bf16_wmma<<<gg, 256, 0, stream>>>(Xb, Wqb, bq, gate, nullptr, Qb, nullptr, 0);
  gemm_bf16_wmma<<<gg, 256, 0, stream>>>(Xb, Wkb, bk, gate, nullptr, Kb, nullptr, 0);
  gemm_bf16_wmma<<<gg, 256, 0, stream>>>(Xb, Wvb, bv, gate, nullptr, Vb, nullptr, 0);

  flash_attn_wmma<<<dim3(SEQ / 64, BATCH * N_HEADS), 128, 0, stream>>>(Qb, Kb, Vb, Ab);

  gemm_bf16_wmma<<<gg, 256, 0, stream>>>(Ab, Wob, bo, gate, x, nullptr, out, 1);
}
